// MLoRAFCN_54752243089559
// MI455X (gfx1250) — matrix operand order repaired and checked by reading, not compiled, not run on hardware
//
#include <hip/hip_runtime.h>
#include <hip/hip_bf16.h>

// ---------------------------------------------------------------------------
// MLoRA FCN on MI455X (gfx1250):
//   out = relu(x @ (kernel + A[idx]@B[idx]) + (bias + db[idx]))
// Folded to a single [131072,256]x[256,128] GEMM.
//
// Roofline: 192 MB moved -> ~8.2us @ 23.3 TB/s; 8.6 GFLOP (AI ~45 FLOP/B)
// => must use matrix pipes. bf16 WMMA (16x16x32) with a 2-term bf16 split of
// both operands (xhi*Whi + xlo*Whi + xhi*Wlo) gives ~f32 accuracy at 3x bf16
// cost, still under the HBM roofline.
//
// Tiling: 64x64 output tile per wave so each LDS B-fragment (hi+lo, 2KB/wave)
// feeds 12 WMMAs -> ~170 B LDS/WMMA, below the ~256 B/clk/WGP LDS supply
// (the previous 16x128 tiling was ~683 B/WMMA, LDS-bound ~2.7x).
// ---------------------------------------------------------------------------

typedef __attribute__((ext_vector_type(16))) __bf16 v16bf;
typedef __attribute__((ext_vector_type(8)))  float  v8f;

#define D_IN   256
#define UNITS  128
#define LORA_R 4
#define BATCH  131072

// ---------------------------------------------------------------------------
// Kernel 1: fold LoRA into the dense weights, emit bf16 hi/lo transposed
// weight planes Wt[N][K] plus fused bias. 32768 threads, trivial cost.
// ---------------------------------------------------------------------------
__global__ __launch_bounds__(256) void mlora_prep(
    const float* __restrict__ a_kernel,     // [64, 256, 4]
    const float* __restrict__ b_kernel,     // [64, 4, 128]
    const float* __restrict__ domain_bias,  // [64, 128]
    const float* __restrict__ kernel,       // [256, 128]
    const float* __restrict__ bias,         // [128]
    const int*   __restrict__ dom,          // [B,1] -> dom[0]
    __bf16* __restrict__ wt_hi,             // [128, 256] (N-major)
    __bf16* __restrict__ wt_lo,             // [128, 256]
    float*  __restrict__ bprime)            // [128]
{
    const int idx = dom[0];
    const int t = blockIdx.x * 256 + threadIdx.x;   // 0 .. 32767
    const int k = t >> 7;                           // 0 .. 255
    const int n = t & 127;                          // 0 .. 127

    const float* A  = a_kernel + (size_t)idx * D_IN * LORA_R + k * LORA_R;
    const float* Bm = b_kernel + (size_t)idx * LORA_R * UNITS;

    float w = kernel[k * UNITS + n];
#pragma unroll
    for (int r = 0; r < LORA_R; ++r)
        w += A[r] * Bm[r * UNITS + n];

    const __bf16 hi = (__bf16)w;
    const __bf16 lo = (__bf16)(w - (float)hi);
    wt_hi[n * D_IN + k] = hi;   // transposed: contiguous along K per column n
    wt_lo[n * D_IN + k] = lo;

    if (t < UNITS)
        bprime[t] = bias[t] + domain_bias[(size_t)idx * UNITS + t];
}

// ---------------------------------------------------------------------------
// Kernel 2: main GEMM. 8 waves/block; each wave computes a 64(M)x64(N) tile:
//   4 M-tiles x 4 N-tiles of f32 accumulators (128 VGPRs),
//   K loop: 8 steps of 32; per step: 4 A-frag hi/lo pairs from global x,
//   4 B-frag hi/lo pairs from LDS, 48 WMMAs.
// Wt_hi/Wt_lo (128 KB) staged once into LDS per block (320 KB/WGP available).
// ---------------------------------------------------------------------------
__global__ __launch_bounds__(256) void mlora_gemm(
    const float*  __restrict__ x,       // [131072, 256]
    const __bf16* __restrict__ wt_hi,   // [128, 256]
    const __bf16* __restrict__ wt_lo,   // [128, 256]
    const float*  __restrict__ bprime,  // [128]
    float*        __restrict__ out)     // [131072, 128]
{
    extern __shared__ __bf16 lds[];     // [2][128*256] bf16 = 128 KB
    __bf16* __restrict__ lhi = lds;
    __bf16* __restrict__ llo = lds + UNITS * D_IN;

    // --- cooperative stage of both weight planes into LDS (b128 traffic) ---
    {
        const uint4* sh = (const uint4*)wt_hi;
        const uint4* sl = (const uint4*)wt_lo;
        uint4* dh = (uint4*)lhi;
        uint4* dl = (uint4*)llo;
        const int n16 = (UNITS * D_IN * 2) / 16;    // 4096 uint4 per plane
        for (int i = threadIdx.x; i < n16; i += 256) {
            dh[i] = sh[i];
            dl[i] = sl[i];
        }
    }
    __syncthreads();

    const int wave   = threadIdx.x >> 5;
    const int lane   = threadIdx.x & 31;
    const int half   = lane >> 4;        // 0: lanes 0-15, 1: lanes 16-31
    const int lane16 = lane & 15;

    // wave task: 64-row strip, 64-col half. Column-half pairs share x rows
    // within the block, so the second read of each x row hits WGP$/L2.
    const int  chalf = wave & 1;                         // 0 or 1
    const long rbase = ((long)blockIdx.x * 4 + (wave >> 1)) * 64;
    const int  cbase = chalf * 64;

    // x row pointers for the 4 M-tiles (A-layout half offset baked in)
    const float* __restrict__ xr[4];
#pragma unroll
    for (int mt = 0; mt < 4; ++mt)
        xr[mt] = x + (rbase + mt * 16 + lane16) * (long)D_IN + half * 16;

    v8f acc[4][4];
#pragma unroll
    for (int mt = 0; mt < 4; ++mt)
#pragma unroll
        for (int nt = 0; nt < 4; ++nt) acc[mt][nt] = (v8f)0.0f;

    for (int s = 0; s < 8; ++s) {        // K loop: 8 steps of 32
        const int k0 = s * 32;

        // ---- A fragments: contiguous k order (same as B below)
        //  lanes 0-15 : K = k0+{0..15}
        //  lanes 16-31: K = k0+{16..31}   (half*16 in pointer)
        v16bf ah[4], al[4];
#pragma unroll
        for (int mt = 0; mt < 4; ++mt) {
            const float4 c0a = *(const float4*)(xr[mt] + k0);
            const float4 c0b = *(const float4*)(xr[mt] + k0 + 4);
            const float4 c1a = *(const float4*)(xr[mt] + k0 + 8);
            const float4 c1b = *(const float4*)(xr[mt] + k0 + 12);
            const float fa[16] = { c0a.x, c0a.y, c0a.z, c0a.w,
                                   c0b.x, c0b.y, c0b.z, c0b.w,
                                   c1a.x, c1a.y, c1a.z, c1a.w,
                                   c1b.x, c1b.y, c1b.z, c1b.w };
#pragma unroll
            for (int i = 0; i < 16; ++i) {
                const __bf16 h = (__bf16)fa[i];
                ah[mt][i] = h;
                al[mt][i] = (__bf16)(fa[i] - (float)h);  // exact residual
            }
        }

#pragma unroll
        for (int nt = 0; nt < 4; ++nt) {
            // ---- B fragment: column n = cbase+nt*16+lane16; lanes 0-15 hold
            // K=k0..k0+15, lanes 16-31 K=k0+16..k0+31 (contiguous in LDS)
            const int boff = (cbase + nt * 16 + lane16) * D_IN + k0 + half * 16;
            const v16bf bh = *(const v16bf*)(lhi + boff);
            const v16bf bl = *(const v16bf*)(llo + boff);

#pragma unroll
            for (int mt = 0; mt < 4; ++mt) {
                acc[mt][nt] = __builtin_amdgcn_wmma_f32_16x16x32_bf16(
                    false, ah[mt], false, bh, (short)0, acc[mt][nt], false, false);
                acc[mt][nt] = __builtin_amdgcn_wmma_f32_16x16x32_bf16(
                    false, al[mt], false, bh, (short)0, acc[mt][nt], false, false);
                acc[mt][nt] = __builtin_amdgcn_wmma_f32_16x16x32_bf16(
                    false, ah[mt], false, bl, (short)0, acc[mt][nt], false, false);
            }
        }
    }

    // ---- epilogue: fused bias + relu, stored per f32 C/D layout:
    //  VGPR v: lanes 0-15 -> row v, lanes 16-31 -> row v+8; col = lane16+...
    //  Non-temporal stores: output is streaming, keep x resident in L2.
#pragma unroll
    for (int nt = 0; nt < 4; ++nt) {
        const int col = cbase + nt * 16 + lane16;
        const float bb = bprime[col];
#pragma unroll
        for (int mt = 0; mt < 4; ++mt) {
            const long r0 = rbase + mt * 16 + half * 8;
#pragma unroll
            for (int v = 0; v < 8; ++v) {
                const float val = acc[mt][nt][v] + bb;
                __builtin_nontemporal_store(val > 0.0f ? val : 0.0f,
                                            &out[(r0 + v) * UNITS + col]);
            }
        }
    }
}

// ---------------------------------------------------------------------------
extern "C" void kernel_launch(void* const* d_in, const int* in_sizes, int n_in,
                              void* d_out, int out_size, void* d_ws, size_t ws_size,
                              hipStream_t stream) {
    const float* x            = (const float*)d_in[0];
    const int*   domain       = (const int*)  d_in[1];
    const float* a_kernel     = (const float*)d_in[2];
    const float* b_kernel     = (const float*)d_in[3];
    const float* domain_bias  = (const float*)d_in[4];
    const float* kernel       = (const float*)d_in[5];
    const float* bias         = (const float*)d_in[6];
    float* out = (float*)d_out;

    char* ws = (char*)d_ws;
    __bf16* wt_hi  = (__bf16*)(ws);                 // 64 KB
    __bf16* wt_lo  = (__bf16*)(ws + 64 * 1024);     // 64 KB
    float*  bprime = (float*) (ws + 128 * 1024);    // 512 B

    // Fold LoRA into dense weights (32768 elements -> 128 blocks x 256).
    mlora_prep<<<128, 256, 0, stream>>>(a_kernel, b_kernel, domain_bias,
                                        kernel, bias, domain,
                                        wt_hi, wt_lo, bprime);

    // Main GEMM: 2048 row-strips x 2 col-halves = 4096 wave tasks,
    // 8 waves/block -> 512 blocks.
    const int smem = 2 * UNITS * D_IN * (int)sizeof(__bf16);   // 128 KB
    mlora_gemm<<<512, 256, smem, stream>>>(x, wt_hi, wt_lo, bprime, out);
}
